// READ_GRU_87763361727029
// MI455X (gfx1250) — matrix operand order repaired and checked by reading, not compiled, not run on hardware
//
#include <hip/hip_runtime.h>
#include <hip/hip_fp16.h>

typedef __attribute__((ext_vector_type(16))) _Float16 v16h;
typedef __attribute__((ext_vector_type(8)))  _Float16 v8h;
typedef __attribute__((ext_vector_type(8)))  float    v8f;
typedef __attribute__((ext_vector_type(4)))  float    v4f;

constexpr int    kL    = 13;
constexpr int    kB    = 16;
constexpr int    kS    = 512;
constexpr int    kH    = 768;
constexpr int    kHD   = 32;
constexpr int    kNT   = kB * kS;        // 8192 tokens
constexpr int    kNCOL = 192;            // fw 96 cols + bw 96 cols
constexpr size_t kLBH  = (size_t)kB * kS * kH;   // per-layer stride in hidden_states

// workspace layout (bytes)
constexpr size_t OFF_XP     = 0;                                   // [13][8192][192] f32
constexpr size_t XP_BYTES   = (size_t)kL * kNT * kNCOL * 4;        // 81,788,928
constexpr size_t OFF_WF16   = OFF_XP + XP_BYTES;                   // [192][768] f16 col-major
constexpr size_t WF16_BYTES = (size_t)kNCOL * kH * 2;              // 294,912
constexpr size_t OFF_HSUM   = OFF_WF16 + WF16_BYTES;               // [8192][32] f32
constexpr size_t HSUM_BYTES = (size_t)kNT * kHD * 4;               // 1,048,576
constexpr size_t OFF_POOL   = OFF_HSUM + HSUM_BYTES;               // [16][768] f32

__device__ __forceinline__ float sigmoidf_(float x) {
    return 1.f / (1.f + __expf(-x));
}
__device__ __forceinline__ float tanhf_(float x) {
    float e = __expf(-2.f * fabsf(x));
    float r = (1.f - e) / (1.f + e);
    return copysignf(r, x);
}

// ---------------------------------------------------------------------------
// Kernel 0: pack Wi_f||Wi_b -> f16 column-major [col][k] and zero pooled accum
// ---------------------------------------------------------------------------
__global__ void __launch_bounds__(256)
prep_kernel(const float* __restrict__ Wi_f, const float* __restrict__ Wi_b,
            _Float16* __restrict__ Wf16, float* __restrict__ pooled) {
    int idx = blockIdx.x * 256 + threadIdx.x;          // 0 .. 147455
    if (idx < kNCOL * kH) {
        int col = idx / kH;
        int k   = idx % kH;
        float v = (col < 96) ? Wi_f[(size_t)k * 96 + col]
                             : Wi_b[(size_t)k * 96 + (col - 96)];
        Wf16[(size_t)col * kH + k] = (_Float16)v;
    }
    if (idx < kB * kH) pooled[idx] = 0.f;
}

// ---------------------------------------------------------------------------
// Kernel 1: XP[t][n][col] = x_t @ [Wi_f|Wi_b] + bias   via WMMA f16->f32
//   grid = (128 token-tiles of 64, 13 layers), block = 128 (4 waves)
//   wave: one 16-token M tile x all 12 N tiles, K = 768 in 24 chunks of 32
// ---------------------------------------------------------------------------
__global__ void __launch_bounds__(128)
proj_kernel(const float* __restrict__ HS, const _Float16* __restrict__ Wf16,
            const float* __restrict__ b_f, const float* __restrict__ b_b,
            float* __restrict__ XP) {
    const int t    = blockIdx.y;
    const int wave = threadIdx.x >> 5;
    const int lane = threadIdx.x & 31;
    const int m0   = (blockIdx.x * 4 + wave) * 16;     // first token of tile
    const int mrow = lane & 15;
    const int hi   = lane >> 4;                        // lane group 0/1

    const float* Xt = HS + (size_t)t * kLBH + (size_t)m0 * kH;
    // A layout (16-bit 16x32): lo lanes hold K {0..7,16..23}, hi lanes +8
    const float* arow = Xt + (size_t)mrow * kH + hi * 8;

    v8f acc[12];
#pragma unroll
    for (int j = 0; j < 12; ++j)
        acc[j] = v8f{0.f, 0.f, 0.f, 0.f, 0.f, 0.f, 0.f, 0.f};

    for (int kc = 0; kc < 24; ++kc) {
        const float* ap = arow + kc * 32;
        __builtin_prefetch(ap + 32, 0, 3);             // global_prefetch_b8 next K chunk
        v4f a0 = *(const v4f*)(ap);
        v4f a1 = *(const v4f*)(ap + 4);
        v4f a2 = *(const v4f*)(ap + 16);
        v4f a3 = *(const v4f*)(ap + 20);
        v16h afrag;
#pragma unroll
        for (int i = 0; i < 4; ++i) {
            afrag[i]      = (_Float16)a0[i];
            afrag[4 + i]  = (_Float16)a1[i];
            afrag[8 + i]  = (_Float16)a2[i];
            afrag[12 + i] = (_Float16)a3[i];
        }
        // B layout (16-bit 32x16): lane = col within tile; lo lanes K {0..7,16..23}, hi lanes +8
        const _Float16* bbase = Wf16 + (size_t)mrow * kH + kc * 32 + hi * 8;
#pragma unroll
        for (int j = 0; j < 12; ++j) {
            const v8h* bp = (const v8h*)(bbase + (size_t)j * 16 * kH);
            v8h blo = bp[0];
            v8h bhi = bp[2];
            v16h bfrag;
#pragma unroll
            for (int i = 0; i < 8; ++i) { bfrag[i] = blo[i]; bfrag[8 + i] = bhi[i]; }
            acc[j] = __builtin_amdgcn_wmma_f32_16x16x32_f16(
                false, afrag, false, bfrag, (short)0, acc[j], false, false);
        }
    }

    // C layout: VGPR r, lanes 0-15 -> M=r, lanes 16-31 -> M=8+r; N = lane&15
#pragma unroll
    for (int j = 0; j < 12; ++j) {
        int col    = j * 16 + mrow;
        float bias = (col < 96) ? b_f[col] : b_b[col - 96];   // input bias b[0]
        float* op  = XP + ((size_t)t * kNT + m0 + 8 * hi) * kNCOL + col;
#pragma unroll
        for (int r = 0; r < 8; ++r)
            op[(size_t)r * kNCOL] = acc[j][r] + bias;
    }
}

// ---------------------------------------------------------------------------
// Kernel 2: GRU recurrence, wave-per-token (wave32), fw then bw, h in registers
// ---------------------------------------------------------------------------
__device__ __forceinline__ float gru_step(float h, const float* __restrict__ xt,
                                          const float* __restrict__ U,
                                          float bz, float br, float bh, int lane) {
    float xz = xt[lane], xr = xt[32 + lane], xh = xt[64 + lane];
    float rz = bz, rr = br, rh = bh;
#pragma unroll
    for (int k = 0; k < kHD; ++k) {
        float hk = __shfl(h, k, 32);
        const float* ur = U + k * 96;
        rz += hk * ur[lane];
        rr += hk * ur[32 + lane];
        rh += hk * ur[64 + lane];
    }
    float z  = sigmoidf_(xz + rz);
    float r  = sigmoidf_(xr + rr);
    float hh = tanhf_(xh + r * rh);
    return z * h + (1.f - z) * hh;
}

__global__ void __launch_bounds__(128)
gru_kernel(const float* __restrict__ XP,
           const float* __restrict__ Uh_f, const float* __restrict__ Uh_b,
           const float* __restrict__ b_f,  const float* __restrict__ b_b,
           float* __restrict__ hsum) {
    __shared__ float Uf[kHD * 96];
    __shared__ float Ub[kHD * 96];
    for (int i = threadIdx.x; i < kHD * 96; i += 128) {
        Uf[i] = Uh_f[i];
        Ub[i] = Uh_b[i];
    }
    __syncthreads();

    const int lane = threadIdx.x & 31;
    const int n    = blockIdx.x * 4 + (threadIdx.x >> 5);

    // recurrent bias b[1]
    float bzf = b_f[96 + lane], brf = b_f[128 + lane], bhf = b_f[160 + lane];
    float bzb = b_b[96 + lane], brb = b_b[128 + lane], bhb = b_b[160 + lane];

    float hf = 0.f;
    for (int t = 0; t < kL; ++t)
        hf = gru_step(hf, XP + ((size_t)t * kNT + n) * kNCOL, Uf, bzf, brf, bhf, lane);

    float hb = 0.f;
    for (int t = kL - 1; t >= 0; --t)
        hb = gru_step(hb, XP + ((size_t)t * kNT + n) * kNCOL + 96, Ub, bzb, brb, bhb, lane);

    hsum[(size_t)n * kHD + lane] = hf + hb;
}

// ---------------------------------------------------------------------------
// Kernel 3: corrected = HS[-1] + tanh(hsum @ W_lin + b_lin); partial pooling
//   grid = (16 b, 16 s-chunks of 32, 3 h-chunks of 256), block = 256
// ---------------------------------------------------------------------------
__global__ void __launch_bounds__(256)
corr_kernel(const float* __restrict__ HS, const float* __restrict__ hsum,
            const float* __restrict__ W_lin, const float* __restrict__ b_lin,
            float* __restrict__ pooled) {
    const int b   = blockIdx.x;
    const int sc  = blockIdx.y;
    const int hc  = blockIdx.z;
    const int tid = threadIdx.x;
    const int h   = hc * 256 + tid;

    __shared__ float wl[kHD * 256];
    __shared__ float hv[kHD];
#pragma unroll
    for (int k = 0; k < kHD; ++k)
        wl[k * 256 + tid] = W_lin[(size_t)k * kH + h];   // each thread reads its own slot back

    const float* hs12 = HS + (size_t)12 * kLBH + ((size_t)b * kS + sc * 32) * kH + h;
    const float  bl   = b_lin[h];
    float acc = 0.f;

    for (int s = 0; s < 32; ++s) {
        __syncthreads();
        if (tid < kHD)
            hv[tid] = hsum[(size_t)(b * kS + sc * 32 + s) * kHD + tid];
        __syncthreads();
        float dot = bl;
#pragma unroll
        for (int k = 0; k < kHD; ++k)
            dot += hv[k] * wl[k * 256 + tid];
        acc += hs12[(size_t)s * kH] + tanhf_(dot);
    }
    atomicAdd(&pooled[b * kH + h], acc);
}

// ---------------------------------------------------------------------------
// Kernel 4: logits = (pooled/512) @ W_cls + b_cls; softmax over 3 classes
// ---------------------------------------------------------------------------
__global__ void __launch_bounds__(64)
head_kernel(const float* __restrict__ pooled, const float* __restrict__ W_cls,
            const float* __restrict__ b_cls, float* __restrict__ out) {
    __shared__ float lg[kB * 3];
    int tid = threadIdx.x;
    if (tid < kB * 3) {
        int b = tid / 3, c = tid % 3;
        float acc = b_cls[c];
        for (int h = 0; h < kH; ++h)
            acc += (pooled[(size_t)b * kH + h] * (1.f / (float)kS)) * W_cls[(size_t)h * 3 + c];
        lg[tid] = acc;
    }
    __syncthreads();
    if (tid < kB) {
        float a = lg[tid * 3], bb = lg[tid * 3 + 1], cc = lg[tid * 3 + 2];
        float m  = fmaxf(a, fmaxf(bb, cc));
        float ea = __expf(a - m), eb = __expf(bb - m), ec = __expf(cc - m);
        float inv = 1.f / (ea + eb + ec);
        out[tid * 3 + 0] = ea * inv;
        out[tid * 3 + 1] = eb * inv;
        out[tid * 3 + 2] = ec * inv;
    }
}

// ---------------------------------------------------------------------------
extern "C" void kernel_launch(void* const* d_in, const int* in_sizes, int n_in,
                              void* d_out, int out_size, void* d_ws, size_t ws_size,
                              hipStream_t stream) {
    const float* HS    = (const float*)d_in[0];
    const float* Wi_f  = (const float*)d_in[1];
    const float* Uh_f  = (const float*)d_in[2];
    const float* b_f   = (const float*)d_in[3];
    const float* Wi_b  = (const float*)d_in[4];
    const float* Uh_b  = (const float*)d_in[5];
    const float* b_b   = (const float*)d_in[6];
    const float* W_lin = (const float*)d_in[7];
    const float* b_lin = (const float*)d_in[8];
    const float* W_cls = (const float*)d_in[9];
    const float* b_cls = (const float*)d_in[10];
    float* out = (float*)d_out;

    char* ws = (char*)d_ws;
    float*    XP     = (float*)(ws + OFF_XP);
    _Float16* Wf16   = (_Float16*)(ws + OFF_WF16);
    float*    hsum   = (float*)(ws + OFF_HSUM);
    float*    pooled = (float*)(ws + OFF_POOL);

    prep_kernel<<<(kNCOL * kH + 255) / 256, 256, 0, stream>>>(Wi_f, Wi_b, Wf16, pooled);
    proj_kernel<<<dim3(kNT / 64, kL), 128, 0, stream>>>(HS, Wf16, b_f, b_b, XP);
    gru_kernel<<<kNT / 4, 128, 0, stream>>>(XP, Uh_f, Uh_b, b_f, b_b, hsum);
    corr_kernel<<<dim3(kB, 16, 3), 256, 0, stream>>>(HS, hsum, W_lin, b_lin, pooled);
    head_kernel<<<1, 64, 0, stream>>>(pooled, W_cls, b_cls, out);
}
